// EdgeModel_52029233824512
// MI455X (gfx1250) — matrix operand order repaired and checked by reading, not compiled, not run on hardware
//
#include <hip/hip_runtime.h>
#include <stdint.h>

#define N_NODES 50000
#define N_EDGES 1600000
#define NODE_IN 128
#define HID 64
#define NODE_TILES (N_NODES / 16)   // 3125 (exact)
#define EDGE_TILES (N_EDGES / 16)   // 100000 (exact)

typedef __attribute__((ext_vector_type(16))) __bf16        v16bf;
typedef __attribute__((ext_vector_type(8)))  float         v8f;
typedef __attribute__((ext_vector_type(8)))  unsigned int  v8u;

// ---------- bf16 helpers (round-to-nearest-even) ----------
__device__ __forceinline__ unsigned bf16bits(float f) {
  unsigned u = __float_as_uint(f);
  unsigned r = (u + 0x7FFFu + ((u >> 16) & 1u)) >> 16;
  return r & 0xFFFFu;
}
__device__ __forceinline__ unsigned pkbf(float lo, float hi) {
  return bf16bits(lo) | (bf16bits(hi) << 16);
}
// 16-bit A-matrix 16x32 layout (ISA 7.12.2): VGPR v, lane-half hl -> K base
__device__ __forceinline__ int a_kbase(int ks, int v, int hl) {
  return 32 * ks + ((v < 4) ? 0 : 16) + hl * 8 + (v & 3) * 2;
}
// B fragment from LDS: [frag][lane][8 dwords] -> 2x ds_load_b128 per lane
__device__ __forceinline__ v16bf lds_bfrag(const unsigned* p, int frag, int lane) {
  v8u bu = *(const v8u*)(p + ((size_t)frag * 32 + lane) * 8);
  return __builtin_bit_cast(v16bf, bu);
}

// ---------- repack a [K x 64] f32 weight into B-fragment order ----------
// out layout: [frag = nt*nks+ks][lane(32)][vgpr(8)] packed bf16 pairs.
// Dense 16-bit B 32x16: lane = N + 16*h, VGPR v holds K = (v<4?0:16) + 8h + 2(v&3), +1.
__global__ void k_pack_w(const float* __restrict__ W, int K, int nks,
                         unsigned* __restrict__ out) {
  int idx = blockIdx.x * blockDim.x + threadIdx.x;
  int total = 4 * nks * 32 * 8;
  if (idx >= total) return;
  int v    = idx & 7;
  int lane = (idx >> 3) & 31;
  int rest = idx >> 8;          // frag index = nt*nks + ks
  int ks   = rest % nks;
  int nt   = rest / nks;
  int n  = 16 * nt + (lane & 15);
  int k0 = 32 * ks + ((v < 4) ? 0 : 16) + ((lane >= 16) ? 8 : 0) + 2 * (v & 3);
  float a = (k0     < K) ? W[(size_t)k0       * HID + n] : 0.0f;
  float b = (k0 + 1 < K) ? W[(size_t)(k0 + 1) * HID + n] : 0.0f;
  out[idx] = pkbf(a, b);
}

// ---------- degree count ----------
__global__ void k_deg(const long long* __restrict__ dst, float* __restrict__ deg) {
  int e = blockIdx.x * blockDim.x + threadIdx.x;
  if (e < N_EDGES) atomicAdd(&deg[dst[e]], 1.0f);
}

// ---------- scatter-add of F-wide f32 rows: agg[dst] += feat[src] ----------
template <int F>
__global__ void k_scatter(const long long* __restrict__ src,
                          const long long* __restrict__ dst,
                          const float* __restrict__ feat,
                          float* __restrict__ agg) {
  constexpr int C = F / 4;  // float4 chunks per edge
  long long idx = (long long)blockIdx.x * blockDim.x + threadIdx.x;
  long long e = idx / C;
  int c = (int)(idx - e * C);
  if (e >= N_EDGES) return;
  long long s = src[e], d = dst[e];
  const float4 val = *(const float4*)(feat + s * F + c * 4);
  float* a = agg + d * F + (size_t)c * 4;
  atomicAdd(a + 0, val.x);
  atomicAdd(a + 1, val.y);
  atomicAdd(a + 2, val.z);
  atomicAdd(a + 3, val.w);
}

// ---------- SAGE layer: out = relu( (agg/deg) @ Wl + b + self @ Wr ) ----------
// One wave per 16-node tile; bf16 WMMA with f32 accumulate; weights in LDS.
template <int KIN, bool OUTBF>
__global__ void k_sage(const float* __restrict__ agg, const float* __restrict__ deg,
                       const float* __restrict__ self,
                       const unsigned* __restrict__ Bl, const unsigned* __restrict__ Br,
                       const float* __restrict__ bias,
                       float* __restrict__ outf, unsigned short* __restrict__ outb) {
  constexpr int NKS = KIN / 32;
  constexpr int NFRAG = 4 * NKS;
  __shared__ unsigned sBl[NFRAG * 256];
  __shared__ unsigned sBr[NFRAG * 256];
  for (int i = threadIdx.x; i < NFRAG * 256; i += blockDim.x) {
    sBl[i] = Bl[i];
    sBr[i] = Br[i];
  }
  __syncthreads();

  int lane = threadIdx.x & 31;
  int tile = blockIdx.x * (blockDim.x >> 5) + (threadIdx.x >> 5);
  if (tile >= NODE_TILES) return;  // uniform per wave -> EXEC stays all-ones
  int hl = lane >> 4;
  int m  = lane & 15;
  int row = tile * 16 + m;
  float inv = 1.0f / fmaxf(deg[row], 1.0f);

  v16bf Am[NKS], As[NKS];
#pragma unroll
  for (int ks = 0; ks < NKS; ++ks) {
    v8u au, su;
#pragma unroll
    for (int v = 0; v < 8; ++v) {
      int k = a_kbase(ks, v, hl);
      au[v] = pkbf(agg[(size_t)row * KIN + k] * inv,
                   agg[(size_t)row * KIN + k + 1] * inv);
      su[v] = pkbf(self[(size_t)row * KIN + k],
                   self[(size_t)row * KIN + k + 1]);
    }
    Am[ks] = __builtin_bit_cast(v16bf, au);
    As[ks] = __builtin_bit_cast(v16bf, su);
  }

#pragma unroll
  for (int nt = 0; nt < 4; ++nt) {
    v8f c = {};
#pragma unroll
    for (int ks = 0; ks < NKS; ++ks) {
      v16bf B = lds_bfrag(sBl, nt * NKS + ks, lane);
      c = __builtin_amdgcn_wmma_f32_16x16x32_bf16(false, Am[ks], false, B,
                                                  (short)0, c, false, false);
    }
#pragma unroll
    for (int ks = 0; ks < NKS; ++ks) {
      v16bf B = lds_bfrag(sBr, nt * NKS + ks, lane);
      c = __builtin_amdgcn_wmma_f32_16x16x32_bf16(false, As[ks], false, B,
                                                  (short)0, c, false, false);
    }
    float bb = bias[nt * 16 + m];
#pragma unroll
    for (int r = 0; r < 8; ++r) {
      float h = fmaxf(c[r] + bb, 0.0f);
      // C/D layout: lane half hl holds rows r + 8*hl, column = lane&15 within N-tile
      size_t orow = (size_t)tile * 16 + r + 8 * hl;
      size_t ocol = (size_t)nt * 16 + m;
      if (OUTBF) outb[orow * HID + ocol] = (unsigned short)bf16bits(h);
      else       outf[orow * HID + ocol] = h;
    }
  }
}

// ---------- edge MLP: sigmoid( relu([h[s],h[d],ea] @ Wm1 + bm1) @ Wm2 + bm2 ) ----------
// K = 144 padded to 160 -> 5 bf16 WMMA k-steps, 4 N-tiles.
// Weights staged in LDS (20 KB per block, amortized over 16 waves/tiles).
// One 16-edge tile per wave; NO per-wave loop so LICM cannot hoist the
// LDS B reads into long-lived registers (which caused scratch spills).
__global__ void k_edge(const long long* __restrict__ src,
                       const long long* __restrict__ dst,
                       const unsigned* __restrict__ h2u,   // bf16 pairs, [N_NODES][32] dwords
                       const float* __restrict__ ea,
                       const unsigned* __restrict__ Bm1,
                       const float* __restrict__ bm1,
                       const float* __restrict__ Wm2,
                       const float* __restrict__ bm2,
                       float* __restrict__ out) {
  constexpr int NKS = 5;
  constexpr int NFRAG = 4 * NKS;
  __shared__ unsigned sB[NFRAG * 256];   // 20 KB
  for (int i = threadIdx.x; i < NFRAG * 256; i += blockDim.x) sB[i] = Bm1[i];
  __syncthreads();

  int lane = threadIdx.x & 31;
  int hl = lane >> 4;
  int m  = lane & 15;
  int tile = blockIdx.x * (blockDim.x >> 5) + (threadIdx.x >> 5);
  if (tile >= EDGE_TILES) return;  // uniform per wave

  long long e = (long long)tile * 16 + m;
  long long s = src[e], d = dst[e];

  v16bf A[NKS];
#pragma unroll
  for (int ks = 0; ks < NKS; ++ks) {
    v8u au;
#pragma unroll
    for (int v = 0; v < 8; ++v) {
      int k0 = a_kbase(ks, v, hl);
      unsigned p;
      if      (k0 < 64)  p = h2u[s * 32 + (k0 >> 1)];
      else if (k0 < 128) p = h2u[d * 32 + ((k0 - 64) >> 1)];
      else if (k0 < 144) p = pkbf(ea[e * 16 + (k0 - 128)], ea[e * 16 + (k0 - 127)]);
      else               p = 0u;  // K padding 144..159
      au[v] = p;
    }
    A[ks] = __builtin_bit_cast(v16bf, au);
  }

  float part[8];
#pragma unroll
  for (int r = 0; r < 8; ++r) part[r] = 0.0f;

#pragma unroll
  for (int nt = 0; nt < 4; ++nt) {
    v8f c = {};
#pragma unroll
    for (int ks = 0; ks < NKS; ++ks) {
      v16bf B = lds_bfrag(sB, nt * NKS + ks, lane);
      c = __builtin_amdgcn_wmma_f32_16x16x32_bf16(false, A[ks], false, B,
                                                  (short)0, c, false, false);
    }
    float b1 = bm1[nt * 16 + m];
    float w2 = Wm2[nt * 16 + m];
#pragma unroll
    for (int r = 0; r < 8; ++r)
      part[r] += fmaxf(c[r] + b1, 0.0f) * w2;
  }

  // reduce across the 16 lanes of each half (rows 0-7 in lanes 0-15, 8-15 in 16-31)
#pragma unroll
  for (int off = 1; off < 16; off <<= 1)
#pragma unroll
    for (int r = 0; r < 8; ++r)
      part[r] += __shfl_xor(part[r], off, 32);

  if (m == 0) {
    float bo = bm2[0];
#pragma unroll
    for (int r = 0; r < 8; ++r) {
      float z = part[r] + bo;
      out[(size_t)tile * 16 + 8 * hl + r] = 1.0f / (1.0f + expf(-z));
    }
  }
}

extern "C" void kernel_launch(void* const* d_in, const int* in_sizes, int n_in,
                              void* d_out, int out_size, void* d_ws, size_t ws_size,
                              hipStream_t stream) {
  (void)in_sizes; (void)n_in; (void)out_size; (void)ws_size;
  const float* x   = (const float*)d_in[0];
  const float* ea  = (const float*)d_in[1];
  const float* W1l = (const float*)d_in[2];
  const float* b1l = (const float*)d_in[3];
  const float* W1r = (const float*)d_in[4];
  const float* W2l = (const float*)d_in[5];
  const float* b2l = (const float*)d_in[6];
  const float* W2r = (const float*)d_in[7];
  const float* Wm1 = (const float*)d_in[8];
  const float* bm1 = (const float*)d_in[9];
  const float* Wm2 = (const float*)d_in[10];
  const float* bm2 = (const float*)d_in[11];
  const long long* ei  = (const long long*)d_in[12];  // int64 [2, E]
  const long long* src = ei;
  const long long* dst = ei + N_EDGES;
  float* out = (float*)d_out;

  // workspace carve-out (256B aligned)
  char* base = (char*)d_ws;
  size_t off = 0;
  auto take = [&](size_t bytes) -> void* {
    void* p = base + off;
    off = (off + bytes + 255) & ~(size_t)255;
    return p;
  };
  float*          agg1 = (float*)take((size_t)N_NODES * NODE_IN * 4);   // 25.6 MB
  float*          deg  = (float*)take((size_t)N_NODES * 4);             // 0.2 MB
  float*          h1   = (float*)take((size_t)N_NODES * HID * 4);       // 12.8 MB
  float*          agg2 = (float*)take((size_t)N_NODES * HID * 4);       // 12.8 MB
  unsigned short* h2   = (unsigned short*)take((size_t)N_NODES * HID * 2); // 6.4 MB
  unsigned* pW1l = (unsigned*)take((size_t)4 * 4 * 8 * 32 * 4);
  unsigned* pW1r = (unsigned*)take((size_t)4 * 4 * 8 * 32 * 4);
  unsigned* pW2l = (unsigned*)take((size_t)4 * 2 * 8 * 32 * 4);
  unsigned* pW2r = (unsigned*)take((size_t)4 * 2 * 8 * 32 * 4);
  unsigned* pWm1 = (unsigned*)take((size_t)4 * 5 * 8 * 32 * 4);

  hipMemsetAsync(agg1, 0, (size_t)N_NODES * NODE_IN * 4, stream);
  hipMemsetAsync(deg,  0, (size_t)N_NODES * 4, stream);
  hipMemsetAsync(agg2, 0, (size_t)N_NODES * HID * 4, stream);

  // repack weights into WMMA B-fragment order (tiny, L2-hot afterwards)
  k_pack_w<<<(4 * 4 * 256) / 256, 256, 0, stream>>>(W1l, 128, 4, pW1l);
  k_pack_w<<<(4 * 4 * 256) / 256, 256, 0, stream>>>(W1r, 128, 4, pW1r);
  k_pack_w<<<(4 * 2 * 256) / 256, 256, 0, stream>>>(W2l, 64, 2, pW2l);
  k_pack_w<<<(4 * 2 * 256) / 256, 256, 0, stream>>>(W2r, 64, 2, pW2r);
  k_pack_w<<<(4 * 5 * 256) / 256, 256, 0, stream>>>(Wm1, 144, 5, pWm1);

  // layer 1: scatter-mean + SAGE
  k_deg<<<N_EDGES / 256, 256, 0, stream>>>(dst, deg);
  k_scatter<128><<<(N_EDGES * 32) / 256, 256, 0, stream>>>(src, dst, x, agg1);
  k_sage<128, false><<<(NODE_TILES + 7) / 8, 256, 0, stream>>>(
      agg1, deg, x, pW1l, pW1r, b1l, h1, (unsigned short*)nullptr);

  // layer 2: scatter-mean + SAGE (output stored bf16 for the edge gathers)
  k_scatter<64><<<(N_EDGES * 16) / 256, 256, 0, stream>>>(src, dst, h1, agg2);
  k_sage<64, true><<<(NODE_TILES + 7) / 8, 256, 0, stream>>>(
      agg2, deg, h1, pW2l, pW2r, b2l, (float*)nullptr, h2);

  // edge MLP + sigmoid: one tile per wave, 16 waves/block share the LDS weights
  k_edge<<<EDGE_TILES / 16, 512, 0, stream>>>(src, dst, (const unsigned*)h2, ea,
                                              pWm1, bm1, Wm2, bm2, out);
}
